// SparseChem_Backbone_64785286692881
// MI455X (gfx1250) — matrix operand order repaired and checked by reading, not compiled, not run on hardware
//
#include <hip/hip_runtime.h>

// ---------------------------------------------------------------------------
// SparseChem backbone: out = relu(relu(X@W_in+b_in ... wait: L1 has no relu)
//   h0 = X  @ W_in + b_in                       [4096 x 4000]
//   h1 = relu(h0 @ W1 + b1)                     [4096 x 4000]
//   y  = relu(h1 @ W2 + b2)                     [4096 x 2000]
// Single tiled WMMA GEMM kernel (f16 inputs converted on the fly, f32 accum),
// launched three times. 128x128 block tile, BK=32, 8 waves, 64x32 wave tile.
// ---------------------------------------------------------------------------

typedef __attribute__((ext_vector_type(16))) _Float16 v16h;
typedef __attribute__((ext_vector_type(8)))  _Float16 v8h;
typedef __attribute__((ext_vector_type(4)))  _Float16 v4h;
typedef __attribute__((ext_vector_type(8)))  float    v8f;
typedef __attribute__((ext_vector_type(4)))  float    v4f;

#define LDH 40  // f16 elements per LDS row (80 bytes: 16B-aligned, bank-friendly)

__global__ __launch_bounds__(256) void gemm_bias_act(
    const float* __restrict__ A,     // [M x K] row-major, fp32
    const float* __restrict__ Bm,    // [K x N] row-major, fp32
    const float* __restrict__ bias,  // [N]
    float* __restrict__ C,           // [M x N]
    int M, int N, int K, int relu)
{
    __shared__ __align__(16) _Float16 ldsA[128 * LDH];  // 128 rows x 32 k (f16)
    __shared__ __align__(16) _Float16 ldsB[128 * LDH];  // 128 n-rows x 32 k (transposed)

    const int t    = threadIdx.x;
    const int lane = t & 31;
    const int w    = t >> 5;        // wave id 0..7
    const int wm   = w >> 2;        // 0..1  (M direction, 64 rows each)
    const int wn   = w & 3;         // 0..3  (N direction, 32 cols each)
    const int lr   = lane & 15;     // lane row/col within 16
    const int kh   = lane >> 4;     // lane-half selects K sub-range

    const int bm = blockIdx.y * 128;
    const int bn = blockIdx.x * 128;

    // -------- global A mapping: thread -> (row, 16-float half-row) ---------
    const int arow = t >> 1;             // 0..127
    const int ac0  = (t & 1) * 16;       // 0 or 16
    const float* Ag = A + (long)(bm + arow) * K + ac0;

    // -------- global B mapping: thread -> (4 k rows, 4 consecutive n) ------
    const int kq = t >> 5;               // 0..7  -> k group of 4
    const int nl = (t & 31) * 4;         // 0..124
    const int ng = bn + nl;
    const bool nin = (ng < N);           // N % 4 == 0, whole float4 in/out

    v4f aR[4], bR[4];
    v8f acc[4][2];
#pragma unroll
    for (int i = 0; i < 4; ++i)
#pragma unroll
        for (int j = 0; j < 2; ++j)
            acc[i][j] = (v8f){0.f, 0.f, 0.f, 0.f, 0.f, 0.f, 0.f, 0.f};

    auto load_tile = [&](int kt) {
#pragma unroll
        for (int i = 0; i < 4; ++i)
            aR[i] = *(const v4f*)(Ag + (long)kt * 32 + i * 4);
#pragma unroll
        for (int i = 0; i < 4; ++i) {
            if (nin) bR[i] = *(const v4f*)(Bm + (long)(kt * 32 + kq * 4 + i) * N + ng);
            else     bR[i] = (v4f){0.f, 0.f, 0.f, 0.f};
        }
    };

    auto store_tile = [&]() {
        // A: 16 consecutive f16 into row arow at col ac0 (two b128 stores)
        v8f lo = __builtin_shufflevector(aR[0], aR[1], 0, 1, 2, 3, 4, 5, 6, 7);
        v8f hi = __builtin_shufflevector(aR[2], aR[3], 0, 1, 2, 3, 4, 5, 6, 7);
        *(v8h*)&ldsA[arow * LDH + ac0]     = __builtin_convertvector(lo, v8h);
        *(v8h*)&ldsA[arow * LDH + ac0 + 8] = __builtin_convertvector(hi, v8h);
        // B: transpose 4k x 4n micro-block; k becomes contiguous (b64 stores)
#pragma unroll
        for (int j = 0; j < 4; ++j) {
            v4h col;
            col[0] = (_Float16)bR[0][j];
            col[1] = (_Float16)bR[1][j];
            col[2] = (_Float16)bR[2][j];
            col[3] = (_Float16)bR[3][j];
            *(v4h*)&ldsB[(nl + j) * LDH + kq * 4] = col;
        }
    };

    auto compute_tile = [&]() {
        v16h af[4], bf[2];
        // A fragment (ISA 16-bit A 16x32 layout): lane<16 -> K0..7,K16..23;
        // lane>=16 -> K8..15,K24..31  => two b128 loads at +kh*8 and +16+kh*8
#pragma unroll
        for (int i = 0; i < 4; ++i) {
            const _Float16* p = &ldsA[(wm * 64 + i * 16 + lr) * LDH + kh * 8];
            v8h lo = *(const v8h*)p;
            v8h hi = *(const v8h*)(p + 16);
            af[i] = __builtin_shufflevector(lo, hi, 0, 1, 2, 3, 4, 5, 6, 7,
                                            8, 9, 10, 11, 12, 13, 14, 15);
        }
        // B fragment: lane<16 -> K0..7,K16..23 of column n; lane>=16 -> K8..15,K24..31
#pragma unroll
        for (int j = 0; j < 2; ++j) {
            const _Float16* p = &ldsB[(wn * 32 + j * 16 + lr) * LDH + kh * 8];
            v8h lo = *(const v8h*)p;
            v8h hi = *(const v8h*)(p + 16);
            bf[j] = __builtin_shufflevector(lo, hi, 0, 1, 2, 3, 4, 5, 6, 7,
                                            8, 9, 10, 11, 12, 13, 14, 15);
        }
#pragma unroll
        for (int i = 0; i < 4; ++i)
#pragma unroll
            for (int j = 0; j < 2; ++j)
                acc[i][j] = __builtin_amdgcn_wmma_f32_16x16x32_f16(
                    false, af[i], false, bf[j], (short)0, acc[i][j], false, false);
    };

    const int KT = K >> 5;  // K % 32 == 0 for all three layers
    load_tile(0);
    for (int kt = 0; kt < KT; ++kt) {
        __syncthreads();            // previous tile fully consumed
        store_tile();
        __syncthreads();            // LDS tile visible
        if (kt + 1 < KT) load_tile(kt + 1);   // overlap next loads with WMMA
        compute_tile();
    }

    // -------- epilogue: bias + optional ReLU, fp32 store -------------------
    // C/D layout: lane<16 -> N=lane, M = r; lane>=16 -> N=lane-16, M = r+8
#pragma unroll
    for (int i = 0; i < 4; ++i) {
        const int m = bm + wm * 64 + i * 16 + kh * 8;
#pragma unroll
        for (int j = 0; j < 2; ++j) {
            const int n = bn + wn * 32 + j * 16 + lr;
            if (n < N) {
                const float bb = bias[n];
#pragma unroll
                for (int r = 0; r < 8; ++r) {
                    float v = acc[i][j][r] + bb;
                    if (relu) v = v > 0.f ? v : 0.f;
                    C[(long)(m + r) * N + n] = v;
                }
            }
        }
    }
}

extern "C" void kernel_launch(void* const* d_in, const int* in_sizes, int n_in,
                              void* d_out, int out_size, void* d_ws, size_t ws_size,
                              hipStream_t stream) {
    const float* X   = (const float*)d_in[0];
    const float* Win = (const float*)d_in[1];
    const float* bin = (const float*)d_in[2];
    const float* W1  = (const float*)d_in[3];
    const float* b1  = (const float*)d_in[4];
    const float* W2  = (const float*)d_in[5];
    const float* b2  = (const float*)d_in[6];
    float* out = (float*)d_out;

    const int B = 4096, D = 32000, H0 = 4000, H1 = 4000, T = 2000;
    float* h0 = (float*)d_ws;                 // [B x H0] fp32
    float* h1 = h0 + (size_t)B * H0;          // [B x H1] fp32

    dim3 blk(256);
    // L1: no ReLU (SparseLinear + bias only)
    gemm_bias_act<<<dim3((H0 + 127) / 128, B / 128), blk, 0, stream>>>(
        X, Win, bin, h0, B, H0, D, 0);
    // L2: Linear + ReLU
    gemm_bias_act<<<dim3((H1 + 127) / 128, B / 128), blk, 0, stream>>>(
        h0, W1, b1, h1, B, H1, H0, 1);
    // L3 (tail): Linear + ReLU -> d_out
    gemm_bias_act<<<dim3((T + 127) / 128, B / 128), blk, 0, stream>>>(
        h1, W2, b2, out, B, T, H1, 1);
}
